// ElasticAST_20907900797213
// MI455X (gfx1250) — compile-verified, matrix operand order checked by reading, not run on hardware
//
#include <hip/hip_runtime.h>
#include <hip/hip_bf16.h>
#include <math.h>

// ---------------------------------------------------------------------------
// ElasticAST forward for MI455X (gfx1250): bf16 WMMA GEMMs, async global->LDS
// DMA for activation tiles, masked block-diagonal attention with in-LDS
// softmax, attention pooling.
// ---------------------------------------------------------------------------

typedef __attribute__((ext_vector_type(16))) __bf16 v16bf;
typedef __attribute__((ext_vector_type(8)))  __bf16 v8bf;
typedef __attribute__((ext_vector_type(8)))  float  v8f;

#define DM    768
#define NTOK  1024
#define NHEAD 12
#define DHEAD 64

static __device__ __forceinline__ v8f wmma_bf16(v16bf a, v16bf b, v8f c) {
    return __builtin_amdgcn_wmma_f32_16x16x32_bf16(false, a, false, b, (short)0, c,
                                                   false, false);
}

static __device__ __forceinline__ v16bf cat8(v8bf lo, v8bf hi) {
    return __builtin_shufflevector(lo, hi, 0, 1, 2, 3, 4, 5, 6, 7,
                                           8, 9, 10, 11, 12, 13, 14, 15);
}

// ---------------------------------------------------------------------------
// GEMM: out[M,Ncols] = act(A[M,K](bf16) @ W[K,Ncols](f32) + bias) (+resid)
// 256 threads = 8 waves; block tile 64(M) x 128(N); wave tile 16x64
// (4 accumulators sharing one A fragment). A tile staged via gfx1250
// global_load_async_to_lds_b128 DMA; W tile converted f32->bf16 and stored
// transposed so fragments are contiguous 16B ds_load_b128 reads.
// ---------------------------------------------------------------------------
template<int ACT, bool RESID, bool WF32, bool WBF16>
__global__ __launch_bounds__(256)
void gemm_wmma_kernel(const __bf16* __restrict__ A, int Mrows, int K,
                      const float* __restrict__ W, int Ncols,
                      const float* __restrict__ bias,
                      const float* __restrict__ resid,
                      float* __restrict__ outF, __bf16* __restrict__ outB) {
    __shared__ __bf16 As[64][32];     // 4 KB
    __shared__ __bf16 BsT[128][40];   // 10 KB, transposed, +8 pad (keeps 16B align)

    const int tid  = threadIdx.x;
    const int lane = tid & 31;
    const int wave = tid >> 5;
    const int half = lane >> 4;
    const int ln16 = lane & 15;
    const int wm   = wave >> 1;       // 0..3 : 16-row subtile
    const int wn   = wave & 1;        // 0..1 : 64-col half
    const int mtile = blockIdx.y * 64;
    const int ntile = blockIdx.x * 128;

    // per-thread async-DMA slot: 256 threads x 16B == 64x32 bf16 tile
    const int arow = tid >> 2;
    const int aseg = (tid & 3) * 8;
    const unsigned a_lds = (unsigned)(unsigned long long)&As[arow][aseg];
    int ag_row = mtile + arow;
    if (ag_row >= Mrows) ag_row = Mrows - 1;   // clamp (stores are guarded)
    const __bf16* arow_ptr = A + (size_t)ag_row * K + aseg;

    v8f acc[4] = {{}, {}, {}, {}};
    for (int kk = 0; kk < K; kk += 32) {
        // ---- A tile: async global -> LDS DMA (no VGPR staging) ----
        {
            const __bf16* ap = arow_ptr + kk;
            asm volatile("global_load_async_to_lds_b128 %0, %1, off"
                         :: "v"(a_lds), "v"(ap) : "memory");
        }
        // ---- W tile: 32x128 f32 -> bf16, stored transposed ----
        for (int idx = tid; idx < 32 * 128; idx += 256) {
            int r = idx >> 7, c = idx & 127;
            int n = ntile + c;
            float w = (n < Ncols) ? W[(size_t)(kk + r) * Ncols + n] : 0.0f;
            BsT[c][r] = (__bf16)w;
        }
        if (kk + 32 < K) {
            __builtin_prefetch(&W[(size_t)(kk + 32 + (tid >> 7)) * Ncols +
                                  ntile + (tid & 127)], 0, 1);
        }
        asm volatile("s_wait_asynccnt 0" ::: "memory");
        __syncthreads();

        v16bf af;
        {
            v8bf lo = *(const v8bf*)&As[wm * 16 + ln16][half * 8];
            v8bf hi = *(const v8bf*)&As[wm * 16 + ln16][16 + half * 8];
            af = cat8(lo, hi);
        }
#pragma unroll
        for (int t = 0; t < 4; ++t) {
            const int col = wn * 64 + t * 16 + ln16;
            v8bf lo = *(const v8bf*)&BsT[col][half * 8];
            v8bf hi = *(const v8bf*)&BsT[col][16 + half * 8];
            acc[t] = wmma_bf16(af, cat8(lo, hi), acc[t]);
        }
        __syncthreads();
    }

#pragma unroll
    for (int t = 0; t < 4; ++t) {
        const int col = ntile + wn * 64 + t * 16 + ln16;
        if (col < Ncols) {
#pragma unroll
            for (int r = 0; r < 8; ++r) {
                const int row = mtile + wm * 16 + r + 8 * half;
                if (row < Mrows) {
                    float v = acc[t][r] + bias[col];
                    if (ACT == 1) v = 0.5f * v * (1.0f + erff(v * 0.70710678118654752f));
                    if (RESID)    v += resid[(size_t)row * Ncols + col];
                    if (WF32)     outF[(size_t)row * Ncols + col] = v;
                    if (WBF16)    outB[(size_t)row * Ncols + col] = (__bf16)v;
                }
            }
        }
    }
}

// ---------------------------------------------------------------------------
// Masked attention for one (b, head, 16-query tile).
// Scores (16 x 1024 f32) live in LDS (64 KB); no 1/sqrt(d) scaling (faithful).
// Block: 128 threads = 4 waves.
// ---------------------------------------------------------------------------
__global__ __launch_bounds__(128)
void attn_kernel(const __bf16* __restrict__ q, const __bf16* __restrict__ kv,
                 const int* __restrict__ sid, const unsigned char* __restrict__ pad,
                 __bf16* __restrict__ out) {
    __shared__ float sc[16][NTOK];   // 64 KB

    const int tid  = threadIdx.x;
    const int lane = tid & 31;
    const int wave = tid >> 5;
    const int half = lane >> 4;
    const int ln16 = lane & 15;
    const int qb = blockIdx.x * 16;
    const int h  = blockIdx.y;
    const int b  = blockIdx.z;
    const size_t rowq = (size_t)b * NTOK + qb;
    const int D2 = 2 * DM;

    // hoist the 8 query-row sample ids this lane masks against
    int sq[8];
#pragma unroll
    for (int r = 0; r < 8; ++r) sq[r] = sid[b * NTOK + qb + r + 8 * half];

    // ---- scores: S = Q @ K^T, masked ----
    for (int jb = wave * 16; jb < NTOK; jb += 64) {
        v8f acc = {};
#pragma unroll
        for (int s = 0; s < 2; ++s) {
            const int kb = s * 32;
            const __bf16* qrow = q  + (rowq + ln16) * DM + h * DHEAD + kb;
            const __bf16* krow = kv + ((size_t)b * NTOK + jb + ln16) * D2 + h * DHEAD + kb;
            v16bf af, bfr;
#pragma unroll
            for (int e = 0; e < 8; ++e) {
                af[e]      = qrow[half * 8 + e];
                af[e + 8]  = qrow[16 + half * 8 + e];
                bfr[e]     = krow[half * 8 + e];
                bfr[e + 8] = krow[16 + half * 8 + e];
            }
            acc = wmma_bf16(af, bfr, acc);
        }
        const int j  = jb + ln16;
        const int sj = sid[b * NTOK + j];
        const bool pj = pad[b * NTOK + j] != 0;
#pragma unroll
        for (int r = 0; r < 8; ++r) {
            bool ok = pj && (sq[r] == sj);
            sc[r + 8 * half][j] = ok ? acc[r] : -3.0e38f;
        }
    }
    __syncthreads();

    // ---- row softmax over 1024 (one wave per row, 4 rows per wave) ----
    for (int m = wave; m < 16; m += 4) {
        float mx = -3.4e38f;
        for (int j = lane; j < NTOK; j += 32) mx = fmaxf(mx, sc[m][j]);
        for (int o = 16; o; o >>= 1) mx = fmaxf(mx, __shfl_xor(mx, o, 32));
        float sum = 0.0f;
        for (int j = lane; j < NTOK; j += 32) {
            float e = __expf(sc[m][j] - mx);
            sc[m][j] = e;
            sum += e;
        }
        for (int o = 16; o; o >>= 1) sum += __shfl_xor(sum, o, 32);
        float inv = 1.0f / sum;
        for (int j = lane; j < NTOK; j += 32) sc[m][j] *= inv;
    }
    __syncthreads();

    // ---- O = P @ V ; wave owns a 16-wide d-slice of the 64-dim head ----
    v8f acc = {};
    const size_t vbase = (size_t)b * NTOK;
    const int vcol = DM + h * DHEAD + wave * 16 + ln16;
    for (int kb = 0; kb < NTOK; kb += 32) {
        v16bf af, bfr;
#pragma unroll
        for (int e = 0; e < 8; ++e) {
            af[e]      = (__bf16)sc[ln16][kb + half * 8 + e];
            af[e + 8]  = (__bf16)sc[ln16][kb + 16 + half * 8 + e];
            bfr[e]     = kv[(vbase + kb + half * 8 + e) * D2 + vcol];
            bfr[e + 8] = kv[(vbase + kb + 16 + half * 8 + e) * D2 + vcol];
        }
        acc = wmma_bf16(af, bfr, acc);
    }
#pragma unroll
    for (int r = 0; r < 8; ++r)
        out[(rowq + r + 8 * half) * DM + h * DHEAD + wave * 16 + ln16] = (__bf16)acc[r];
}

// ---------------------------------------------------------------------------
// LayerNorm over last dim (D), one block per row, writes bf16.
// ---------------------------------------------------------------------------
__global__ __launch_bounds__(256)
void ln_kernel(const float* __restrict__ x, const float* __restrict__ g,
               const float* __restrict__ bta, __bf16* __restrict__ out, int D) {
    __shared__ float r1[8], r2[8];
    const int row = blockIdx.x, tid = threadIdx.x;
    const float* xr = x + (size_t)row * D;

    float s = 0.0f;
    for (int d = tid; d < D; d += 256) s += xr[d];
    for (int o = 16; o; o >>= 1) s += __shfl_xor(s, o, 32);
    if ((tid & 31) == 0) r1[tid >> 5] = s;
    __syncthreads();
    float mean = 0.0f;
    for (int i = 0; i < 8; ++i) mean += r1[i];
    mean /= (float)D;

    float v = 0.0f;
    for (int d = tid; d < D; d += 256) { float t = xr[d] - mean; v += t * t; }
    for (int o = 16; o; o >>= 1) v += __shfl_xor(v, o, 32);
    if ((tid & 31) == 0) r2[tid >> 5] = v;
    __syncthreads();
    float var = 0.0f;
    for (int i = 0; i < 8; ++i) var += r2[i];
    var /= (float)D;

    const float inv = rsqrtf(var + 1e-6f);
    for (int d = tid; d < D; d += 256)
        out[(size_t)row * D + d] = (__bf16)((xr[d] - mean) * inv * g[d] + bta[d]);
}

// ---------------------------------------------------------------------------
// Small helpers
// ---------------------------------------------------------------------------
__global__ void cvt_bf16_kernel(const float* __restrict__ in, __bf16* __restrict__ out, int n) {
    int i = blockIdx.x * 256 + threadIdx.x;
    if (i < n) out[i] = (__bf16)in[i];
}

__global__ void addpos_kernel(float* __restrict__ x, const float* __restrict__ pf,
                              const float* __restrict__ pt, const int* __restrict__ fi,
                              const int* __restrict__ ti) {
    const int row = blockIdx.x;
    const int f = fi[row], t = ti[row];
    for (int d = threadIdx.x; d < DM; d += 256)
        x[(size_t)row * DM + d] += pf[(size_t)f * DM + d] + pt[(size_t)t * DM + d];
}

__global__ void qproj_kernel(const float* __restrict__ pq, const float* __restrict__ Wq,
                             const float* __restrict__ bq, float* __restrict__ qout) {
    int c = blockIdx.x * 256 + threadIdx.x;
    if (c < DM) {
        float s = bq[c];
        for (int d = 0; d < DM; ++d) s += pq[d] * Wq[(size_t)d * DM + c];
        qout[c] = s;
    }
}

// Attention pooling: one block per (m, b); M=8 queries share the q vector,
// differ only in the mask sample_ids == m.
__global__ __launch_bounds__(256)
void pool_attn_kernel(const float* __restrict__ qp, const float* __restrict__ kvp,
                      const int* __restrict__ sid, const unsigned char* __restrict__ pad,
                      float* __restrict__ outp) {
    __shared__ float a[NTOK];
    __shared__ float red[256];
    __shared__ float qs[DHEAD];
    const int tid = threadIdx.x;
    const int m = blockIdx.x, b = blockIdx.y;
    const int D2 = 2 * DM;

    for (int h = 0; h < NHEAD; ++h) {
        if (tid < DHEAD) qs[tid] = qp[h * DHEAD + tid];
        __syncthreads();

        for (int j = tid; j < NTOK; j += 256) {
            const float* krow = kvp + ((size_t)b * NTOK + j) * D2 + h * DHEAD;
            float s = 0.0f;
            for (int d = 0; d < DHEAD; ++d) s += qs[d] * krow[d];
            bool ok = (sid[b * NTOK + j] == m) && (pad[b * NTOK + j] != 0);
            a[j] = ok ? s : -3.0e38f;
        }
        __syncthreads();

        float mx = fmaxf(fmaxf(a[tid], a[tid + 256]), fmaxf(a[tid + 512], a[tid + 768]));
        red[tid] = mx; __syncthreads();
        for (int o = 128; o; o >>= 1) { if (tid < o) red[tid] = fmaxf(red[tid], red[tid + o]); __syncthreads(); }
        mx = red[0]; __syncthreads();

        float ls = 0.0f;
        for (int j = tid; j < NTOK; j += 256) { float e = __expf(a[j] - mx); a[j] = e; ls += e; }
        red[tid] = ls; __syncthreads();
        for (int o = 128; o; o >>= 1) { if (tid < o) red[tid] += red[tid + o]; __syncthreads(); }
        const float inv = 1.0f / red[0]; __syncthreads();

        const int d = tid & 63, gsub = tid >> 6;
        float p = 0.0f;
        for (int j = gsub; j < NTOK; j += 4)
            p += a[j] * kvp[((size_t)b * NTOK + j) * D2 + DM + h * DHEAD + d];
        red[tid] = p; __syncthreads();
        if (gsub == 0)
            outp[((size_t)b * 8 + m) * DM + h * DHEAD + d] =
                (red[d] + red[64 + d] + red[128 + d] + red[192 + d]) * inv;
        __syncthreads();
    }
}

// ---------------------------------------------------------------------------
// Host driver
// ---------------------------------------------------------------------------
extern "C" void kernel_launch(void* const* d_in, const int* in_sizes, int n_in,
                              void* d_out, int out_size, void* d_ws, size_t ws_size,
                              hipStream_t stream) {
    (void)in_sizes; (void)n_in; (void)out_size; (void)ws_size;

    const float*         patches    = (const float*)d_in[0];
    const int*           sample_ids = (const int*)d_in[1];
    const unsigned char* key_pad    = (const unsigned char*)d_in[2];
    const int*           freq_idx   = (const int*)d_in[3];
    const int*           time_idx   = (const int*)d_in[4];
    const float* patch_W = (const float*)d_in[5];
    const float* patch_b = (const float*)d_in[6];
    const float* pos_f   = (const float*)d_in[7];
    const float* pos_t   = (const float*)d_in[8];
    const float* ln1_g   = (const float*)d_in[9];
    const float* ln1_b   = (const float*)d_in[10];
    const float* Wq      = (const float*)d_in[11];
    const float* bq      = (const float*)d_in[12];
    const float* Wkv     = (const float*)d_in[13];
    const float* bkv     = (const float*)d_in[14];
    const float* Wo      = (const float*)d_in[15];
    const float* bo      = (const float*)d_in[16];
    const float* ln2_g   = (const float*)d_in[17];
    const float* ln2_b   = (const float*)d_in[18];
    const float* W1      = (const float*)d_in[19];
    const float* b1      = (const float*)d_in[20];
    const float* W2      = (const float*)d_in[21];
    const float* b2      = (const float*)d_in[22];
    const float* norm_g  = (const float*)d_in[23];
    const float* norm_b  = (const float*)d_in[24];
    const float* pool_q  = (const float*)d_in[25];
    const float* pool_Wq = (const float*)d_in[26];
    const float* pool_bq = (const float*)d_in[27];
    const float* pool_Wkv= (const float*)d_in[28];
    const float* pool_bkv= (const float*)d_in[29];
    const float* pool_Wo = (const float*)d_in[30];
    const float* pool_bo = (const float*)d_in[31];
    const float* mln_g   = (const float*)d_in[32];
    const float* mln_b   = (const float*)d_in[33];
    const float* head_W  = (const float*)d_in[34];
    const float* head_b  = (const float*)d_in[35];
    float* out = (float*)d_out;

    const int R = 4 * NTOK;   // 4096 packed rows

    char* ws = (char*)d_ws;
    size_t off = 0;
    auto alloc = [&](size_t bytes) -> void* {
        void* p = ws + off;
        off += (bytes + 255) & ~(size_t)255;
        return p;
    };
    __bf16* pb16   = (__bf16*)alloc((size_t)R * 256 * 2);
    float*  x      = (float*) alloc((size_t)R * DM * 4);
    __bf16* h16    = (__bf16*)alloc((size_t)R * DM * 2);
    __bf16* q16    = (__bf16*)alloc((size_t)R * DM * 2);
    __bf16* kv16   = (__bf16*)alloc((size_t)R * 2 * DM * 2);
    __bf16* a16    = (__bf16*)alloc((size_t)R * DM * 2);
    __bf16* ff16   = (__bf16*)alloc((size_t)R * 3072 * 2);
    __bf16* xn16   = (__bf16*)alloc((size_t)R * DM * 2);
    float*  kvp    = (float*) alloc((size_t)R * 2 * DM * 4);
    float*  qpool  = (float*) alloc(DM * 4);
    float*  ppre   = (float*) alloc((size_t)32 * DM * 4);
    __bf16* pp16   = (__bf16*)alloc((size_t)32 * DM * 2);
    float*  pooled = (float*) alloc((size_t)32 * DM * 4);
    __bf16* pl16   = (__bf16*)alloc((size_t)32 * DM * 2);

    // ---- patch embed + positional embeddings ----
    cvt_bf16_kernel<<<(R * 256 + 255) / 256, 256, 0, stream>>>(patches, pb16, R * 256);
    gemm_wmma_kernel<0, false, true, false><<<dim3(DM / 128, R / 64), 256, 0, stream>>>(
        pb16, R, 256, patch_W, DM, patch_b, nullptr, x, nullptr);
    addpos_kernel<<<R, 256, 0, stream>>>(x, pos_f, pos_t, freq_idx, time_idx);

    // ---- transformer layers ----
    for (int l = 0; l < 12; ++l) {
        ln_kernel<<<R, 256, 0, stream>>>(x, ln1_g + l * DM, ln1_b + l * DM, h16, DM);
        gemm_wmma_kernel<0, false, false, true><<<dim3(DM / 128, R / 64), 256, 0, stream>>>(
            h16, R, DM, Wq + (size_t)l * DM * DM, DM, bq + l * DM, nullptr, nullptr, q16);
        gemm_wmma_kernel<0, false, false, true><<<dim3(2 * DM / 128, R / 64), 256, 0, stream>>>(
            h16, R, DM, Wkv + (size_t)l * DM * 2 * DM, 2 * DM, bkv + l * 2 * DM,
            nullptr, nullptr, kv16);
        attn_kernel<<<dim3(NTOK / 16, NHEAD, 4), 128, 0, stream>>>(
            q16, kv16, sample_ids, key_pad, a16);
        gemm_wmma_kernel<0, true, true, false><<<dim3(DM / 128, R / 64), 256, 0, stream>>>(
            a16, R, DM, Wo + (size_t)l * DM * DM, DM, bo + l * DM, x, x, nullptr);
        ln_kernel<<<R, 256, 0, stream>>>(x, ln2_g + l * DM, ln2_b + l * DM, h16, DM);
        gemm_wmma_kernel<1, false, false, true><<<dim3(3072 / 128, R / 64), 256, 0, stream>>>(
            h16, R, DM, W1 + (size_t)l * DM * 3072, 3072, b1 + l * 3072,
            nullptr, nullptr, ff16);
        gemm_wmma_kernel<0, true, true, false><<<dim3(DM / 128, R / 64), 256, 0, stream>>>(
            ff16, R, 3072, W2 + (size_t)l * 3072 * DM, DM, b2 + l * DM, x, x, nullptr);
    }

    // ---- final norm + attention pooling + head ----
    ln_kernel<<<R, 256, 0, stream>>>(x, norm_g, norm_b, xn16, DM);
    gemm_wmma_kernel<0, false, true, false><<<dim3(2 * DM / 128, R / 64), 256, 0, stream>>>(
        xn16, R, DM, pool_Wkv, 2 * DM, pool_bkv, nullptr, kvp, nullptr);
    qproj_kernel<<<3, 256, 0, stream>>>(pool_q, pool_Wq, pool_bq, qpool);
    pool_attn_kernel<<<dim3(8, 4), 256, 0, stream>>>(qpool, kvp, sample_ids, key_pad, ppre);
    cvt_bf16_kernel<<<(32 * DM + 255) / 256, 256, 0, stream>>>(ppre, pp16, 32 * DM);
    gemm_wmma_kernel<0, false, true, false><<<dim3(DM / 128, 1), 256, 0, stream>>>(
        pp16, 32, DM, pool_Wo, DM, pool_bo, nullptr, pooled, nullptr);
    ln_kernel<<<32, 256, 0, stream>>>(pooled, mln_g, mln_b, pl16, DM);
    gemm_wmma_kernel<0, false, true, false><<<dim3((527 + 127) / 128, 1), 256, 0, stream>>>(
        pl16, 32, DM, head_W, 527, head_b, nullptr, out, nullptr);
}
